// AttentionModule_40982577938688
// MI455X (gfx1250) — matrix operand order repaired and checked by reading, not compiled, not run on hardware
//
#include <hip/hip_runtime.h>

typedef __attribute__((ext_vector_type(16))) _Float16 v16h;
typedef __attribute__((ext_vector_type(8)))  float    v8f;
typedef __attribute__((ext_vector_type(4)))  int      v4i_;

#define BATCH 8
#define CDIM  2048
#define NDIM  2048

struct __align__(16) H8 { _Float16 h[8]; };

union Frag { v16h h; float4 f4[2]; };

#if __has_builtin(__builtin_amdgcn_global_load_async_to_lds_b128)
#define HAVE_ASYNC 1
#define ASYNC_B128(gp, sp)                                                   \
  __builtin_amdgcn_global_load_async_to_lds_b128(                            \
      (__attribute__((address_space(1))) v4i_*)(gp),                         \
      (__attribute__((address_space(3))) v4i_*)(sp), 0, 0)
__device__ __forceinline__ void wait_async0() {
#if __has_builtin(__builtin_amdgcn_s_wait_asynccnt)
  __builtin_amdgcn_s_wait_asynccnt(0);
#else
  asm volatile("s_wait_asynccnt 0" ::: "memory");
#endif
}
#endif

// ---------------- f32 -> f16 convert (8 elems / thread) ----------------
__global__ __launch_bounds__(256) void k_convert(const float* __restrict__ x,
                                                 _Float16* __restrict__ xh, int n8) {
  int i = blockIdx.x * blockDim.x + threadIdx.x;
  if (i >= n8) return;
  const float4* p = (const float4*)x + (size_t)i * 2;
  float4 a = p[0], b = p[1];
  H8 o;
  o.h[0] = (_Float16)a.x; o.h[1] = (_Float16)a.y;
  o.h[2] = (_Float16)a.z; o.h[3] = (_Float16)a.w;
  o.h[4] = (_Float16)b.x; o.h[5] = (_Float16)b.y;
  o.h[6] = (_Float16)b.z; o.h[7] = (_Float16)b.w;
  ((H8*)xh)[i] = o;
}

// ---------------- tiled f16 transpose: in (R x Cc) -> out (Cc x R), per batch z ----
__global__ __launch_bounds__(256) void k_transpose_f16(const _Float16* __restrict__ in,
                                                       _Float16* __restrict__ out,
                                                       int R, int Cc) {
  __shared__ _Float16 tile[32][33];
  size_t boff = (size_t)blockIdx.z * R * Cc;
  in += boff; out += boff;
  int c0 = blockIdx.x * 32, r0 = blockIdx.y * 32;
  int tx = threadIdx.x, ty = threadIdx.y;
#pragma unroll
  for (int i = 0; i < 4; i++)
    tile[ty + i * 8][tx] = in[(size_t)(r0 + ty + i * 8) * Cc + c0 + tx];
  __syncthreads();
#pragma unroll
  for (int i = 0; i < 4; i++)
    out[(size_t)(c0 + ty + i * 8) * R + r0 + tx] = tile[tx][ty + i * 8];
}

// ---------------- GEMM: C[M x Nn] = A[M x K] * Bt[Nn x K]^T  (f16 in, f32 out) ----
// 128(M) x 256(N) block tile, BK=32, 8 waves in 2x4 grid, 64x64 per wave,
// double-buffered LDS fed by GLOBAL_LOAD_ASYNC_TO_LDS_B128 when available.
#define LDSK 40   // padded K stride in halves (80B = 5*16B: aligned, conflict-free)
__global__ __launch_bounds__(256)
void k_gemm_nt(const _Float16* __restrict__ A, const _Float16* __restrict__ Bt,
               float* __restrict__ C, int M, int Nn, int K) {
  __shared__ _Float16 As[2][128 * LDSK];   // 2 x 10 KB
  __shared__ _Float16 Bs[2][256 * LDSK];   // 2 x 20 KB

  int bz = blockIdx.z;
  A  += (size_t)bz * M  * K;
  Bt += (size_t)bz * Nn * K;
  C  += (size_t)bz * M  * Nn;

  int bM = blockIdx.y, bN = blockIdx.x;
  int t = threadIdx.x;
  int lane = t & 31, wid = t >> 5;
  int waveM = wid >> 2, waveN = wid & 3;   // 2 x 4 wave grid, 64x64 each
  int lane16 = lane & 15, laneHi = lane >> 4;

  // staging: A tile 128x32 -> 16 f16/thread; B tile 256x32 -> 32 f16/thread
  const _Float16* gA = A  + (size_t)(bM * 128 + (t >> 1)) * K + (t & 1) * 16;
  const _Float16* gB = Bt + (size_t)(bN * 256 + t) * K;
  int sAoff = (t >> 1) * LDSK + (t & 1) * 16;
  int sBoff = t * LDSK;

  v8f acc[4][4];
#pragma unroll
  for (int i = 0; i < 4; i++)
#pragma unroll
    for (int j = 0; j < 4; j++) acc[i][j] = (v8f)0.0f;

  int nk = K >> 5;
  int kbA = laneHi ? 16 : 0;   // A frag: K run  [kb..kb+15]
  int kbB = laneHi ? 16 : 0;   // B frag: K run  [kb..kb+15]

#ifdef HAVE_ASYNC
  // -------- async DMA staging path --------
  {
    ASYNC_B128(gA,      &As[0][sAoff]);
    ASYNC_B128(gA + 8,  &As[0][sAoff + 8]);
    ASYNC_B128(gB,      &Bs[0][sBoff]);
    ASYNC_B128(gB + 8,  &Bs[0][sBoff + 8]);
    ASYNC_B128(gB + 16, &Bs[0][sBoff + 16]);
    ASYNC_B128(gB + 24, &Bs[0][sBoff + 24]);
  }
  wait_async0();
  __syncthreads();

  for (int ks = 0; ks < nk; ks++) {
    int cur = ks & 1;
    if (ks + 1 < nk) {
      int nxt = cur ^ 1;
      const _Float16* ga = gA + (size_t)(ks + 1) * 32;
      const _Float16* gb = gB + (size_t)(ks + 1) * 32;
      ASYNC_B128(ga,      &As[nxt][sAoff]);
      ASYNC_B128(ga + 8,  &As[nxt][sAoff + 8]);
      ASYNC_B128(gb,      &Bs[nxt][sBoff]);
      ASYNC_B128(gb + 8,  &Bs[nxt][sBoff + 8]);
      ASYNC_B128(gb + 16, &Bs[nxt][sBoff + 16]);
      ASYNC_B128(gb + 24, &Bs[nxt][sBoff + 24]);
    }

    Frag bf[4];
#pragma unroll
    for (int jn = 0; jn < 4; jn++) {
      const _Float16* p = &Bs[cur][(waveN * 64 + jn * 16 + lane16) * LDSK];
      bf[jn].f4[0] = *(const float4*)(p + kbB);
      bf[jn].f4[1] = *(const float4*)(p + kbB + 8);
    }
#pragma unroll
    for (int im = 0; im < 4; im++) {
      Frag af;
      const _Float16* p = &As[cur][(waveM * 64 + im * 16 + lane16) * LDSK];
      af.f4[0] = *(const float4*)(p + kbA);
      af.f4[1] = *(const float4*)(p + kbA + 8);
#pragma unroll
      for (int jn = 0; jn < 4; jn++)
        acc[im][jn] = __builtin_amdgcn_wmma_f32_16x16x32_f16(
            false, af.h, false, bf[jn].h, (short)0, acc[im][jn], false, false);
    }

    if (ks + 1 < nk) {
      wait_async0();
      __syncthreads();
    }
  }
#else
  // -------- register staging fallback --------
  float4 ra0, ra1, rb0, rb1, rb2, rb3;
  ra0 = *(const float4*)(gA);
  ra1 = *(const float4*)(gA + 8);
  rb0 = *(const float4*)(gB);
  rb1 = *(const float4*)(gB + 8);
  rb2 = *(const float4*)(gB + 16);
  rb3 = *(const float4*)(gB + 24);
  *(float4*)&As[0][sAoff]     = ra0;
  *(float4*)&As[0][sAoff + 8] = ra1;
  *(float4*)&Bs[0][sBoff]      = rb0;
  *(float4*)&Bs[0][sBoff + 8]  = rb1;
  *(float4*)&Bs[0][sBoff + 16] = rb2;
  *(float4*)&Bs[0][sBoff + 24] = rb3;
  __syncthreads();

  for (int ks = 0; ks < nk; ks++) {
    int cur = ks & 1;
    if (ks + 1 < nk) {
      const _Float16* ga = gA + (size_t)(ks + 1) * 32;
      const _Float16* gb = gB + (size_t)(ks + 1) * 32;
      ra0 = *(const float4*)(ga);
      ra1 = *(const float4*)(ga + 8);
      rb0 = *(const float4*)(gb);
      rb1 = *(const float4*)(gb + 8);
      rb2 = *(const float4*)(gb + 16);
      rb3 = *(const float4*)(gb + 24);
    }

    Frag bf[4];
#pragma unroll
    for (int jn = 0; jn < 4; jn++) {
      const _Float16* p = &Bs[cur][(waveN * 64 + jn * 16 + lane16) * LDSK];
      bf[jn].f4[0] = *(const float4*)(p + kbB);
      bf[jn].f4[1] = *(const float4*)(p + kbB + 8);
    }
#pragma unroll
    for (int im = 0; im < 4; im++) {
      Frag af;
      const _Float16* p = &As[cur][(waveM * 64 + im * 16 + lane16) * LDSK];
      af.f4[0] = *(const float4*)(p + kbA);
      af.f4[1] = *(const float4*)(p + kbA + 8);
#pragma unroll
      for (int jn = 0; jn < 4; jn++)
        acc[im][jn] = __builtin_amdgcn_wmma_f32_16x16x32_f16(
            false, af.h, false, bf[jn].h, (short)0, acc[im][jn], false, false);
    }

    if (ks + 1 < nk) {
      int nxt = cur ^ 1;
      *(float4*)&As[nxt][sAoff]     = ra0;
      *(float4*)&As[nxt][sAoff + 8] = ra1;
      *(float4*)&Bs[nxt][sBoff]      = rb0;
      *(float4*)&Bs[nxt][sBoff + 8]  = rb1;
      *(float4*)&Bs[nxt][sBoff + 16] = rb2;
      *(float4*)&Bs[nxt][sBoff + 24] = rb3;
      __syncthreads();
    }
  }
#endif

  // C/D layout: VGPR r, lanes 0-15 -> M=r, lanes 16-31 -> M=8+r; N = lane%16
#pragma unroll
  for (int im = 0; im < 4; im++) {
    int row0 = bM * 128 + waveM * 64 + im * 16 + laneHi * 8;
#pragma unroll
    for (int jn = 0; jn < 4; jn++) {
      int col = bN * 256 + waveN * 64 + jn * 16 + lane16;
#pragma unroll
      for (int r = 0; r < 8; r++)
        C[(size_t)(row0 + r) * Nn + col] = acc[im][jn][r];
    }
  }
}

// ---------------- row softmax: S (f32, row of NDIM) -> P (f16) ----------------
__global__ __launch_bounds__(256) void k_softmax(const float* __restrict__ S,
                                                 _Float16* __restrict__ P) {
  int n = blockIdx.x, b = blockIdx.y;
  const float* row = S + ((size_t)b * NDIM + n) * NDIM;
  int t = threadIdx.x, lane = t & 31, wid = t >> 5;

  const float4* r4 = (const float4*)row;
  float4 v0 = r4[t * 2], v1 = r4[t * 2 + 1];
  float e[8] = {v0.x, v0.y, v0.z, v0.w, v1.x, v1.y, v1.z, v1.w};

  float m = e[0];
#pragma unroll
  for (int i = 1; i < 8; i++) m = fmaxf(m, e[i]);
#pragma unroll
  for (int off = 16; off > 0; off >>= 1) m = fmaxf(m, __shfl_xor(m, off, 32));
  __shared__ float red[8];
  if (lane == 0) red[wid] = m;
  __syncthreads();
  float m8 = red[0];
#pragma unroll
  for (int i = 1; i < 8; i++) m8 = fmaxf(m8, red[i]);
  __syncthreads();

  float s = 0.0f;
#pragma unroll
  for (int i = 0; i < 8; i++) { e[i] = __expf(e[i] - m8); s += e[i]; }
#pragma unroll
  for (int off = 16; off > 0; off >>= 1) s += __shfl_xor(s, off, 32);
  if (lane == 0) red[wid] = s;
  __syncthreads();
  float stot = 0.0f;
#pragma unroll
  for (int i = 0; i < 8; i++) stot += red[i];
  float inv = 1.0f / stot;

  H8 o;
#pragma unroll
  for (int i = 0; i < 8; i++) o.h[i] = (_Float16)(e[i] * inv);
  ((H8*)(P + ((size_t)b * NDIM + n) * NDIM))[t] = o;
}

// ---------------------------------------------------------------------------
extern "C" void kernel_launch(void* const* d_in, const int* in_sizes, int n_in,
                              void* d_out, int out_size, void* d_ws, size_t ws_size,
                              hipStream_t stream) {
  const float* x = (const float*)d_in[0];
  char* ws = (char*)d_ws;

  const size_t MB = 1024ull * 1024ull;
  _Float16* Xh = (_Float16*)(ws);                // 64 MB: x as f16, (B,C,N)
  _Float16* Xt = (_Float16*)(ws + 64 * MB);      // 64 MB: x^T as f16, (B,N,C)
  float*    S  = (float*)   (ws + 128 * MB);     // 128 MB: scores f32, (B,N,N)
  _Float16* P  = Xt;                             // softmax out reuses Xt region
  _Float16* Pt = (_Float16*)S;                   // attn^T reuses S region
  float* out = (float*)d_out;

  int n8 = BATCH * CDIM * NDIM / 8;
  k_convert<<<dim3(n8 / 256), dim3(256), 0, stream>>>(x, Xh, n8);

  k_transpose_f16<<<dim3(NDIM / 32, CDIM / 32, BATCH), dim3(32, 8), 0, stream>>>(
      Xh, Xt, CDIM, NDIM);

  // S[n][m] = sum_c Xt[n][c] * Xt[m][c]  (Gram matrix)
  k_gemm_nt<<<dim3(NDIM / 256, NDIM / 128, BATCH), dim3(256), 0, stream>>>(
      Xt, Xt, S, NDIM, NDIM, CDIM);

  k_softmax<<<dim3(NDIM, BATCH), dim3(256), 0, stream>>>(S, P);

  k_transpose_f16<<<dim3(NDIM / 32, NDIM / 32, BATCH), dim3(32, 8), 0, stream>>>(
      P, Pt, NDIM, NDIM);

  // out[c][m] = sum_n Xh[c][n] * Pt[m][n]
  k_gemm_nt<<<dim3(NDIM / 256, CDIM / 128, BATCH), dim3(256), 0, stream>>>(
      Xh, Pt, out, CDIM, NDIM, NDIM);
}
